// CondConv_29609504539080
// MI455X (gfx1250) — matrix operand order repaired and checked by reading, not compiled, not run on hardware
//
#include <hip/hip_runtime.h>
#include <hip/hip_bf16.h>

// ---------------------------------------------------------------------------
// CondConv for MI455X (gfx1250, wave32, WMMA bf16 + Tensor Data Mover)
//   B=32, C_IN=256, C_OUT=256, H=W=56, E=8, K=3, PAD=1
// ---------------------------------------------------------------------------

#define B_      32
#define CIN     256
#define COUT    256
#define HW      3136        // 56*56
#define IMW     56
#define E_      8
#define KK_     2304        // CIN * 3 * 3
#define WELEM   589824      // COUT * KK_
#define KSTEPS  72          // KK_ / 32

typedef __attribute__((ext_vector_type(8)))  float        v8f;
typedef __attribute__((ext_vector_type(8)))  __bf16       v8bf;
typedef __attribute__((ext_vector_type(16))) __bf16       v16bf;
typedef __attribute__((ext_vector_type(4)))  unsigned int u32x4;
typedef __attribute__((ext_vector_type(8)))  int          s32x8;
typedef __attribute__((ext_vector_type(4)))  int          s32x4;

// ---------------------------------------------------------------------------
// Kernel 1: GAP -> 1x1 fc -> sigmoid   (one block per sample, 8 waves)
// ---------------------------------------------------------------------------
__global__ __launch_bounds__(256) void routing_kernel(
    const float* __restrict__ x, const float* __restrict__ fcw,
    const float* __restrict__ fcb, float* __restrict__ routing)
{
    const int b    = blockIdx.x;
    const int lane = threadIdx.x & 31;
    const int wid  = threadIdx.x >> 5;
    __shared__ float gap[CIN];

    const float* xb = x + (size_t)b * CIN * HW;
    for (int c = wid; c < CIN; c += 8) {
        const float* xc = xb + (size_t)c * HW;
        float s = 0.f;
        for (int i = lane; i < HW; i += 32) s += xc[i];
        #pragma unroll
        for (int off = 16; off > 0; off >>= 1) s += __shfl_down(s, off, 32);
        if (lane == 0) gap[c] = s * (1.0f / (float)HW);
    }
    __syncthreads();
    if (threadIdx.x < E_) {
        const int e = threadIdx.x;
        float d = fcb[e];
        const float* w = fcw + e * CIN;
        for (int c = 0; c < CIN; ++c) d += gap[c] * w[c];
        routing[b * E_ + e] = 1.0f / (1.0f + __expf(-d));
    }
}

// ---------------------------------------------------------------------------
// Kernel 2: per-sample weight synthesis, f32 -> bf16
//   Wb[b][o][k] = sum_e r[b,e] * W[e,o,k]   (expert weights read once)
// ---------------------------------------------------------------------------
__global__ __launch_bounds__(256) void synth_kernel(
    const float* __restrict__ kw, const float* __restrict__ routing,
    __bf16* __restrict__ Wb)
{
    __shared__ float r[B_ * E_];
    r[threadIdx.x] = routing[threadIdx.x];
    __syncthreads();

    const size_t pos = (size_t)blockIdx.x * 256 + threadIdx.x;
    float w[E_];
    #pragma unroll
    for (int e = 0; e < E_; ++e) w[e] = kw[(size_t)e * WELEM + pos];

    #pragma unroll 4
    for (int b = 0; b < B_; ++b) {
        float a = 0.f;
        #pragma unroll
        for (int e = 0; e < E_; ++e) a += r[b * E_ + e] * w[e];
        Wb[(size_t)b * WELEM + pos] = (__bf16)a;
    }
}

// ---------------------------------------------------------------------------
// Kernel 3: implicit-GEMM conv, one sample per blockIdx.z, 512 threads.
//   D[256 x 128] tile = Wb[256 x 2304] x im2col[2304 x 128], K-steps of 32.
//   16 waves: 8 across M x 2 across N => 8 WMMA accumulators per wave.
//   A-tile: TDM (tensor_load_to_lds, one DMA per K-step, LDS-padded rows).
//   B-tile: per-lane im2col gather, register-pipelined. LDS double-buffered.
// ---------------------------------------------------------------------------
#define MT   256
#define NT   128
#define KS   32
#define LDT  40     // LDS row pitch in bf16 (80 B = 64 B data + 16 B TDM pad)

// Issue one TDM load: 2D tile (32 x 256 elems, 2 B each) from global (row
// pitch KK_ elems) into LDS at lds_off, padding 4 DWORDs every 16 DWORDs.
__device__ __forceinline__ void tdm_load_A(const __bf16* gsrc, unsigned lds_off)
{
    const unsigned long long ga = (unsigned long long)(uintptr_t)gsrc;
    u32x4 g0;
    g0[0] = 1u;                                          // count=1 (valid D#)
    g0[1] = lds_off;                                     // lds_addr (bytes)
    g0[2] = (unsigned)(ga & 0xffffffffu);                // global_addr[31:0]
    g0[3] = (unsigned)((ga >> 32) & 0x01ffffffu)         // global_addr[56:32]
          | (2u << 30);                                  // type=2 ("image")
    s32x8 g1;
    g1[0] = (1 << 16)      // data_size = 1 -> 2 bytes
          | (1 << 20)      // pad_enable
          | (3 << 22)      // pad_interval code 3 -> every 16 DWORDs (64 B)
          | (3 << 25);     // pad_amount  code 3 -> 4 DWORDs (16 B)
    g1[1] = (int)((KK_ & 0xffffu) << 16);                // tensor_dim0[15:0]
    g1[2] = (int)((KK_ >> 16) | ((MT & 0xffff) << 16));  // dim0 hi, tensor_dim1 lo
    g1[3] = (int)((MT >> 16) | (KS << 16));              // dim1 hi, tile_dim0=32
    g1[4] = MT;                                          // tile_dim1=256, tile_dim2=0
    g1[5] = KK_;                                         // tensor_dim0_stride lo32
    g1[6] = 0;                                           // stride hi, dim1_stride lo
    g1[7] = 0;
    const s32x4 z4 = {0, 0, 0, 0};
    const s32x8 z8 = {0, 0, 0, 0, 0, 0, 0, 0};
    __builtin_amdgcn_tensor_load_to_lds(g0, g1, z4, z4, z8, 0);
}

__global__ __launch_bounds__(512) void conv_kernel(
    const float* __restrict__ x, const __bf16* __restrict__ Wb,
    float* __restrict__ out)
{
    const int b  = blockIdx.z;
    const int p0 = blockIdx.x * NT;

    __shared__ __align__(16) __bf16 Asm[2][MT * LDT];
    __shared__ __align__(16) __bf16 Bsm[2][NT * LDT];

    const int t    = threadIdx.x;
    const int lane = t & 31;
    const int wid  = t >> 5;                 // 0..15

    const __bf16* wbase = Wb + (size_t)b * WELEM;
    const float*  xb    = x + (size_t)b * CIN * HW;

    // LDS byte offsets of the two A buffers (flat addr low 32 bits == LDS addr)
    const unsigned ldsA0 = (unsigned)(uintptr_t)(&Asm[0][0]);
    const unsigned ldsA1 = (unsigned)(uintptr_t)(&Asm[1][0]);

    // uniform "am I wave 0" predicate in an SGPR -> scalar branch
    const bool isWave0 = (__builtin_amdgcn_readfirstlane(t) < 32);

    // ---- B-tile gather mapping: thread owns kk = t>>4, pixels nb..nb+7 ----
    const int kk = t >> 4;                   // 0..31
    const int nb = (t & 15) * 8;
    const int pBase = p0 + nb;
    const int y0 = pBase / IMW;
    const int x0 = pBase - y0 * IMW;
    int ci = kk / 9;                         // incremental im2col k-state
    int rr = kk - ci * 9;

    float bReg[8];
    int   bOk[8];

    auto fetchB = [&]() {
        const int kh  = rr / 3;
        const int kw2 = rr - kh * 3;
        const float* base = xb + (size_t)ci * HW;
        int px = x0, sy = y0 + kh - 1;
        int rowoff = sy * IMW;
        #pragma unroll
        for (int j = 0; j < 8; ++j) {
            const int sx = px + kw2 - 1;
            const bool ok = (pBase + j < HW) & ((unsigned)sy < IMW) &
                            ((unsigned)sx < IMW);
            bOk[j]  = ok;
            bReg[j] = ok ? base[rowoff + sx] : 0.f;
            ++px;
            if (px == IMW) { px = 0; ++sy; rowoff += IMW; }
        }
    };
    auto advB = [&]() {                      // gk += 32 = 3*9 + 5
        ci += 3; rr += 5;
        if (rr >= 9) { rr -= 9; ++ci; }
    };
    auto storeB = [&](int buf) {
        #pragma unroll
        for (int j = 0; j < 8; ++j)
            Bsm[buf][(nb + j) * LDT + kk] = (__bf16)(bOk[j] ? bReg[j] : 0.f);
    };

    // ---- per-wave WMMA tile mapping ----
    const int mbase = (wid & 7) * 32;        // 8 waves across M=256
    const int nbase = (wid >> 3) * 64;       // 2 waves across N=128
    const int arow  = lane & 15;
    const int kba   = (lane >> 4) * 8;       // A frag K-sub-base (ISA layout)
    const int kbb   = (lane >> 4) * 8;       // B frag K-base

    v8f acc[2][4];
    #pragma unroll
    for (int i = 0; i < 2; ++i)
        #pragma unroll
        for (int j = 0; j < 4; ++j) acc[i][j] = (v8f)0.f;

    // ---- prologue: stage K-step 0 ----
    if (isWave0) tdm_load_A(wbase, ldsA0);   // DMA A tile 0
    fetchB();
    storeB(0);
    advB();

    // ---- pipelined main loop ----
    for (int kt = 0; kt < KSTEPS; ++kt) {
        const int buf = kt & 1;
        __builtin_amdgcn_s_wait_tensorcnt(0);   // wave0: A DMA done (others: nop)
        __syncthreads();                        // publish A(TDM) + B stores

        const bool more = (kt + 1 < KSTEPS);
        if (more) {
            if (isWave0)
                tdm_load_A(wbase + (kt + 1) * KS, buf ? ldsA0 : ldsA1);
            fetchB();                           // global gathers, no wait yet
        }

        const __bf16* As = &Asm[buf][0];
        const __bf16* Bs = &Bsm[buf][0];
        v16bf afrag[2];
        #pragma unroll
        for (int ms = 0; ms < 2; ++ms) {
            const int m = mbase + ms * 16 + arow;
            v8bf lo = *(const v8bf*)(&As[m * LDT + kba]);
            v8bf hi = *(const v8bf*)(&As[m * LDT + 16 + kba]);
            afrag[ms] = __builtin_shufflevector(lo, hi,
                0,1,2,3,4,5,6,7,8,9,10,11,12,13,14,15);
        }
        #pragma unroll
        for (int ns = 0; ns < 4; ++ns) {
            const int n = nbase + ns * 16 + arow;
            v8bf lo = *(const v8bf*)(&Bs[n * LDT + kbb]);
            v8bf hi = *(const v8bf*)(&Bs[n * LDT + kbb + 16]);
            v16bf bfr = __builtin_shufflevector(lo, hi,
                0,1,2,3,4,5,6,7,8,9,10,11,12,13,14,15);
            #pragma unroll
            for (int ms = 0; ms < 2; ++ms) {
                acc[ms][ns] = __builtin_amdgcn_wmma_f32_16x16x32_bf16(
                    false, afrag[ms], false, bfr,
                    (short)0, acc[ms][ns], false, false);
            }
        }

        if (more) {
            storeB(buf ^ 1);                 // waits B gathers here, after WMMAs
            advB();
        }
    }

    // ---- epilogue: C layout (§7.12.2): VGPR v -> M = v + 8*(lane>=16) ----
    const int ncol = lane & 15;
    const int mh   = (lane >> 4) * 8;
    #pragma unroll
    for (int ms = 0; ms < 2; ++ms) {
        #pragma unroll
        for (int ns = 0; ns < 4; ++ns) {
            const int n = p0 + nbase + ns * 16 + ncol;
            if (n < HW) {
                const int mlo = mbase + ms * 16 + mh;
                float* op = out + ((size_t)b * COUT + mlo) * HW + n;
                #pragma unroll
                for (int v = 0; v < 8; ++v)
                    op[(size_t)v * HW] = acc[ms][ns][v];
            }
        }
    }
}

// ---------------------------------------------------------------------------
// Launch
// ---------------------------------------------------------------------------
extern "C" void kernel_launch(void* const* d_in, const int* in_sizes, int n_in,
                              void* d_out, int out_size, void* d_ws, size_t ws_size,
                              hipStream_t stream) {
    const float* x   = (const float*)d_in[0];   // [32,256,56,56]
    const float* kw  = (const float*)d_in[1];   // [8,256,256,3,3]
    const float* fcw = (const float*)d_in[2];   // [8,256]
    const float* fcb = (const float*)d_in[3];   // [8]
    float* out = (float*)d_out;                 // [32,256,56,56]

    float*  routing = (float*)d_ws;                   // 256 floats
    __bf16* Wb      = (__bf16*)((char*)d_ws + 1024);  // 37.75 MB

    routing_kernel<<<B_, 256, 0, stream>>>(x, fcw, fcb, routing);
    synth_kernel<<<WELEM / 256, 256, 0, stream>>>(kw, routing, Wb);
    conv_kernel<<<dim3((HW + NT - 1) / NT, 1, B_), 512, 0, stream>>>(x, Wb, out);
}
